// GraphSAGE_76149770158505
// MI455X (gfx1250) — matrix operand order repaired and checked by reading, not compiled, not run on hardware
//
#include <hip/hip_runtime.h>
#include <hip/hip_bf16.h>
#include <stdint.h>

#define NN   100000
#define NE   1600000
#define INF  128
#define HIDF 128
#define OUTF 64
#define KCAT 256   // [x | neigh] concatenated K

typedef __attribute__((ext_vector_type(16))) __bf16 v16bf;
typedef __attribute__((ext_vector_type(8)))  float  v8f;

__device__ __forceinline__ uint16_t f2bf(float f) {
    uint32_t u = __float_as_uint(f);
    u += 0x7FFFu + ((u >> 16) & 1u);          // round-to-nearest-even
    return (uint16_t)(u >> 16);
}

// ---------------- utility kernels ----------------
__global__ void k_zero(float* __restrict__ p, int n) {
    int i = blockIdx.x * blockDim.x + threadIdx.x;
    if (i < n) p[i] = 0.0f;
}

__global__ void k_degree(const int* __restrict__ dst, float* __restrict__ deg, int ne) {
    int e = blockIdx.x * blockDim.x + threadIdx.x;
    if (e < ne) atomicAdd(&deg[dst[e]], 1.0f);
}

// one wave (32 lanes) per edge; 4 floats per lane
__global__ void k_scatter_f32(const float* __restrict__ x, const int* __restrict__ src,
                              const int* __restrict__ dst, float* __restrict__ agg, int ne) {
    int e = blockIdx.x * 8 + (threadIdx.x >> 5);
    if (e >= ne) return;
    int lane = threadIdx.x & 31;
    int s = src[e], d = dst[e];
    const float4 v = *(const float4*)(x + (size_t)s * INF + lane * 4);
    float* o = agg + (size_t)d * INF + lane * 4;
    atomicAdd(o + 0, v.x); atomicAdd(o + 1, v.y);
    atomicAdd(o + 2, v.z); atomicAdd(o + 3, v.w);
}

__global__ void k_scatter_bf16(const uint16_t* __restrict__ x, const int* __restrict__ src,
                               const int* __restrict__ dst, float* __restrict__ agg, int ne) {
    int e = blockIdx.x * 8 + (threadIdx.x >> 5);
    if (e >= ne) return;
    int lane = threadIdx.x & 31;
    int s = src[e], d = dst[e];
    const uint2 v = *(const uint2*)(x + (size_t)s * HIDF + lane * 4);
    float* o = agg + (size_t)d * HIDF + lane * 4;
    atomicAdd(o + 0, __uint_as_float(v.x << 16));
    atomicAdd(o + 1, __uint_as_float(v.x & 0xFFFF0000u));
    atomicAdd(o + 2, __uint_as_float(v.y << 16));
    atomicAdd(o + 3, __uint_as_float(v.y & 0xFFFF0000u));
}

// A1[n] = [bf16(x[n]) | bf16(agg[n]/max(deg,1))]
__global__ void k_finalize1(const float* __restrict__ x, const float* __restrict__ agg,
                            const float* __restrict__ deg, uint16_t* __restrict__ A, int n) {
    int i = blockIdx.x * blockDim.x + threadIdx.x;   // node*128 + k
    if (i >= n) return;
    int node = i >> 7, k = i & 127;
    float inv = 1.0f / fmaxf(deg[node], 1.0f);
    A[(size_t)node * KCAT + k]        = f2bf(x[i]);
    A[(size_t)node * KCAT + 128 + k]  = f2bf(agg[i] * inv);
}

// A2[n] = [h_bf16[n] | bf16(agg[n]/max(deg,1))]
__global__ void k_finalize2(const uint16_t* __restrict__ h, const float* __restrict__ agg,
                            const float* __restrict__ deg, uint16_t* __restrict__ A, int n) {
    int i = blockIdx.x * blockDim.x + threadIdx.x;
    if (i >= n) return;
    int node = i >> 7, k = i & 127;
    float inv = 1.0f / fmaxf(deg[node], 1.0f);
    A[(size_t)node * KCAT + k]        = h[i];
    A[(size_t)node * KCAT + 128 + k]  = f2bf(agg[i] * inv);
}

// Wt[col][k] (k contiguous, 256 per row): k<128 -> Wself[k][col], else Wneigh[k-128][col]
__global__ void k_prep_w(const float* __restrict__ Wself, const float* __restrict__ Wneigh,
                         uint16_t* __restrict__ Wt, int cols) {
    int i = blockIdx.x * blockDim.x + threadIdx.x;   // col*256 + k
    if (i >= cols * KCAT) return;
    int col = i >> 8, k = i & 255;
    float v = (k < 128) ? Wself[(size_t)k * cols + col]
                        : Wneigh[(size_t)(k - 128) * cols + col];
    Wt[i] = f2bf(v);
}

// ---------------- WMMA GEMM: out[NN x COLS] = A[NN x 256](bf16) @ Wt^T + bias ----------------
// One wave computes a 32x32 output tile: 2 A fragments x 2 B fragments -> 4 WMMAs/K-step.
template <bool RELU, bool OUTBF16, int COLS>
__device__ __forceinline__ void store_ctile(const v8f& c, int row0, int col, float bv,
                                            void* __restrict__ outp, int half) {
#pragma unroll
    for (int r = 0; r < 8; ++r) {
        float v = c[r] + bv;
        if (RELU) v = fmaxf(v, 0.0f);
        const size_t idx = (size_t)(row0 + r + 8 * half) * COLS + col;
        if (OUTBF16) ((uint16_t*)outp)[idx] = f2bf(v);
        else         ((float*)outp)[idx]    = v;
    }
}

template <int COLS, bool RELU, bool OUTBF16>
__global__ void k_gemm_wmma(const uint16_t* __restrict__ A, const uint16_t* __restrict__ Wt,
                            const float* __restrict__ bias, void* __restrict__ outp) {
    constexpr int CT = COLS / 32;       // 32-wide col tiles
    constexpr int RT = NN / 32;         // 3125 row tiles (exact)
    const int wave = threadIdx.x >> 5;
    const int lane = threadIdx.x & 31;
    int tile = blockIdx.x * 8 + wave;
    if (tile >= RT * CT) return;
    const int row0 = (tile / CT) * 32;
    const int col0 = (tile % CT) * 32;
    const int m    = lane & 15;
    const int half = lane >> 4;

    // A fragment (16x32 bf16): lane m holds K runs [8h,8h+8) and [16+8h,16+8h+8)
    const uint16_t* ar0 = A + (size_t)(row0 + m) * KCAT;
    const uint16_t* ar1 = A + (size_t)(row0 + 16 + m) * KCAT;
    // B fragment (32x16 bf16): lane n=m holds K runs [8h,8h+8) and [16+8h,16+8h+8) of column col
    const uint16_t* wr0 = Wt + (size_t)(col0 + m) * KCAT;
    const uint16_t* wr1 = Wt + (size_t)(col0 + 16 + m) * KCAT;
    __builtin_prefetch(ar0, 0, 1);
    __builtin_prefetch(ar1, 0, 1);
    __builtin_prefetch(wr0, 0, 1);
    __builtin_prefetch(wr1, 0, 1);

    v8f c00 = {}, c01 = {}, c10 = {}, c11 = {};
#pragma unroll
    for (int k0 = 0; k0 < KCAT; k0 += 32) {
        union { v16bf v; uint4 q[2]; } a0, a1, b0, b1;
        a0.q[0] = *(const uint4*)(ar0 + k0 + 8 * half);
        a0.q[1] = *(const uint4*)(ar0 + k0 + 16 + 8 * half);
        a1.q[0] = *(const uint4*)(ar1 + k0 + 8 * half);
        a1.q[1] = *(const uint4*)(ar1 + k0 + 16 + 8 * half);
        b0.q[0] = *(const uint4*)(wr0 + k0 + 8 * half);
        b0.q[1] = *(const uint4*)(wr0 + k0 + 16 + 8 * half);
        b1.q[0] = *(const uint4*)(wr1 + k0 + 8 * half);
        b1.q[1] = *(const uint4*)(wr1 + k0 + 16 + 8 * half);
        c00 = __builtin_amdgcn_wmma_f32_16x16x32_bf16(false, a0.v, false, b0.v, (short)0, c00, false, false);
        c01 = __builtin_amdgcn_wmma_f32_16x16x32_bf16(false, a0.v, false, b1.v, (short)0, c01, false, false);
        c10 = __builtin_amdgcn_wmma_f32_16x16x32_bf16(false, a1.v, false, b0.v, (short)0, c10, false, false);
        c11 = __builtin_amdgcn_wmma_f32_16x16x32_bf16(false, a1.v, false, b1.v, (short)0, c11, false, false);
    }

    // C/D layout: col = lane&15, row = r + 8*(lane>>4)
    const float bv0 = bias[col0 + m];
    const float bv1 = bias[col0 + 16 + m];
    store_ctile<RELU, OUTBF16, COLS>(c00, row0,      col0 + m,      bv0, outp, half);
    store_ctile<RELU, OUTBF16, COLS>(c01, row0,      col0 + 16 + m, bv1, outp, half);
    store_ctile<RELU, OUTBF16, COLS>(c10, row0 + 16, col0 + m,      bv0, outp, half);
    store_ctile<RELU, OUTBF16, COLS>(c11, row0 + 16, col0 + 16 + m, bv1, outp, half);
}

// ---------------- host launcher ----------------
extern "C" void kernel_launch(void* const* d_in, const int* in_sizes, int n_in,
                              void* d_out, int out_size, void* d_ws, size_t ws_size,
                              hipStream_t stream) {
    (void)in_sizes; (void)n_in; (void)out_size; (void)ws_size;
    const float* x    = (const float*)d_in[0];
    const int*   esrc = (const int*)  d_in[1];
    const int*   edst = (const int*)  d_in[2];
    const float* W1s  = (const float*)d_in[3];
    const float* W1n  = (const float*)d_in[4];
    const float* b1   = (const float*)d_in[5];
    const float* W2s  = (const float*)d_in[6];
    const float* W2n  = (const float*)d_in[7];
    const float* b2   = (const float*)d_in[8];
    float* out = (float*)d_out;

    char* ws = (char*)d_ws;
    size_t off = 0;
    auto alloc = [&](size_t bytes) -> void* {
        void* p = ws + off;
        off = (off + bytes + 255) & ~(size_t)255;
        return p;
    };
    float*    deg  = (float*)   alloc((size_t)NN * 4);
    float*    agg  = (float*)   alloc((size_t)NN * INF * 4);
    uint16_t* Abuf = (uint16_t*)alloc((size_t)NN * KCAT * 2);
    uint16_t* hbf  = (uint16_t*)alloc((size_t)NN * HIDF * 2);
    uint16_t* Wt   = (uint16_t*)alloc((size_t)KCAT * HIDF * 2);

    const int nfeat = NN * INF;   // 12.8M

    // ---- layer 1 ----
    k_zero<<<(NN + 255) / 256, 256, 0, stream>>>(deg, NN);
    k_zero<<<(nfeat + 255) / 256, 256, 0, stream>>>(agg, nfeat);
    k_degree<<<(NE + 255) / 256, 256, 0, stream>>>(edst, deg, NE);
    k_scatter_f32<<<(NE + 7) / 8, 256, 0, stream>>>(x, esrc, edst, agg, NE);
    k_prep_w<<<(KCAT * HIDF + 255) / 256, 256, 0, stream>>>(W1s, W1n, Wt, HIDF);
    k_finalize1<<<(nfeat + 255) / 256, 256, 0, stream>>>(x, agg, deg, Abuf, nfeat);
    {
        const int tiles = (NN / 32) * (HIDF / 32);     // 3125 * 4 = 12500
        k_gemm_wmma<HIDF, true, true>
            <<<(tiles + 7) / 8, 256, 0, stream>>>(Abuf, Wt, b1, hbf);
    }

    // ---- layer 2 ----
    k_zero<<<(nfeat + 255) / 256, 256, 0, stream>>>(agg, nfeat);
    k_scatter_bf16<<<(NE + 7) / 8, 256, 0, stream>>>(hbf, esrc, edst, agg, NE);
    k_prep_w<<<(KCAT * OUTF + 255) / 256, 256, 0, stream>>>(W2s, W2n, Wt, OUTF);
    k_finalize2<<<(nfeat + 255) / 256, 256, 0, stream>>>(hbf, agg, deg, Abuf, nfeat);
    {
        const int tiles = (NN / 32) * (OUTF / 32);     // 3125 * 2 = 6250
        k_gemm_wmma<OUTF, false, false>
            <<<(tiles + 7) / 8, 256, 0, stream>>>(Abuf, Wt, b2, out);
    }
}
